// DecoderLayer_66331474920176
// MI455X (gfx1250) — matrix operand order repaired and checked by reading, not compiled, not run on hardware
//
#include <hip/hip_runtime.h>
#include <cmath>

// ---------------------------------------------------------------------------
// MI455X (gfx1250) implementation of the decoder layer.
//  - All GEMMs via v_wmma_f32_16x16x32_f16 (f32 I/O, f16 compute, f32 accum)
//  - Flash-attention (no LxL score tensor) with WMMA for QK^T and PV
//  - Register-staged software pipeline: next tile's global loads overlap
//    current tile's WMMA compute (waits land before the LDS stores)
//  - Serial scans (EMA / cumsum) as one-thread-per-(b,d) loops
// Workspace requirement: 13 slabs * B*L*D floats = ~218 MB.
// ---------------------------------------------------------------------------

typedef __attribute__((ext_vector_type(16))) _Float16 v16h;
typedef __attribute__((ext_vector_type(8)))  _Float16 v8h;
typedef __attribute__((ext_vector_type(8)))  float    v8f;

static __device__ __forceinline__ v16h cat8(v8h a, v8h b) {
  return __builtin_shufflevector(a, b, 0,1,2,3,4,5,6,7,8,9,10,11,12,13,14,15);
}
static __device__ __forceinline__ float dsigmoid(float x) { return 1.0f / (1.0f + __expf(-x)); }
static __device__ __forceinline__ float dgelu(float x)    { return 0.5f * x * (1.0f + erff(x * 0.70710678118f)); }

constexpr int BB   = 4;
constexpr int LL   = 2048;
constexpr int DD   = 512;
constexpr int DFF  = 2048;
constexpr int HH   = 8;
constexpr int DH   = 64;
constexpr int BL   = BB * LL;                 // 8192 rows
constexpr size_t BLD = (size_t)BL * DD;       // 4,194,304 elems per slab

// ===========================================================================
// Generic WMMA GEMM:  C[M,N] = act( A[M,K] @ W[K,N] + bias + add )
//  block = 128 threads (4 waves), tile 64x64, K-step 32.
//  act: 0 none, 1 relu, 2 gelu(erf)
//  Requires M%64==0, N%64==0, K%32==0 (true for every call site here).
// ===========================================================================
#define LDP 40   // padded LDS stride in halfs (80B: 16B aligned, conflict-free)

__global__ void __launch_bounds__(128)
gemm_wmma_kernel(const float* __restrict__ A, const float* __restrict__ W,
                 const float* __restrict__ bias, const float* add,
                 float* C, int M, int N, int K, int act)
{
  __shared__ _Float16 sA[64 * LDP];   // [row][k]
  __shared__ _Float16 sB[64 * LDP];   // [col][k]   (W tile stored transposed)

  const int tid  = threadIdx.x;
  const int m0   = blockIdx.y * 64;
  const int n0   = blockIdx.x * 64;
  const int wave = tid >> 5;
  const int lane = tid & 31;
  const int hi   = lane >> 4;
  const int ln   = lane & 15;
  const int wm   = (wave >> 1) * 32;
  const int wn   = (wave & 1) * 32;

  v8f acc[2][2] = {};

  // register staging buffers for the software pipeline
  float4 ra[4], rb[4];
  auto load_tile = [&](int kt) {
#pragma unroll
    for (int r = 0; r < 4; ++r) {                 // A tile: 64 rows x 32 k
      int i = tid + 128 * r, row = i >> 3, c4 = i & 7;
      ra[r] = *(const float4*)(A + (size_t)(m0 + row) * K + kt + c4 * 4);
    }
#pragma unroll
    for (int r = 0; r < 4; ++r) {                 // W tile: 32 k x 64 cols
      int i = tid + 128 * r, krow = i >> 4, c4 = i & 15;
      rb[r] = *(const float4*)(W + (size_t)(kt + krow) * N + n0 + c4 * 4);
    }
  };

  load_tile(0);
  for (int kt = 0; kt < K; kt += 32) {
    __syncthreads();                              // prior compute done with LDS
    // --- dump staged registers into LDS as f16 ---------------------------
#pragma unroll
    for (int r = 0; r < 4; ++r) {
      int i = tid + 128 * r, row = i >> 3, c4 = i & 7;
      _Float16* d = &sA[row * LDP + c4 * 4];
      d[0] = (_Float16)ra[r].x; d[1] = (_Float16)ra[r].y;
      d[2] = (_Float16)ra[r].z; d[3] = (_Float16)ra[r].w;
    }
#pragma unroll
    for (int r = 0; r < 4; ++r) {
      int i = tid + 128 * r, krow = i >> 4, c4 = i & 15;
      sB[(c4 * 4 + 0) * LDP + krow] = (_Float16)rb[r].x;
      sB[(c4 * 4 + 1) * LDP + krow] = (_Float16)rb[r].y;
      sB[(c4 * 4 + 2) * LDP + krow] = (_Float16)rb[r].z;
      sB[(c4 * 4 + 3) * LDP + krow] = (_Float16)rb[r].w;
    }
    __syncthreads();                              // LDS tile ready
    if (kt + 32 < K) load_tile(kt + 32);          // overlaps with WMMAs below

    // --- build fragments per ISA 7.12.2 layouts ---------------------------
    v16h af[2], bf[2];
#pragma unroll
    for (int i = 0; i < 2; ++i) {
      int m = wm + i * 16 + ln;   // A: lane = M row; K(j)=j+(j>=8?8:0)+hi*8
      v8h a0 = *(const v8h*)&sA[m * LDP + hi * 8];
      v8h a1 = *(const v8h*)&sA[m * LDP + 16 + hi * 8];
      af[i] = cat8(a0, a1);
      int n = wn + i * 16 + ln;   // B: lane = N col; K(j)=j+(j>=8?8:0)+hi*8
      v8h b0 = *(const v8h*)&sB[n * LDP + hi * 8];
      v8h b1 = *(const v8h*)&sB[n * LDP + 16 + hi * 8];
      bf[i] = cat8(b0, b1);
    }
#pragma unroll
    for (int i = 0; i < 2; ++i)
#pragma unroll
      for (int j = 0; j < 2; ++j)
        acc[i][j] = __builtin_amdgcn_wmma_f32_16x16x32_f16(
            false, af[i], false, bf[j], (short)0, acc[i][j], false, false);
  }

  // --- epilogue: C layout lane=N col, vgpr r => row r + hi*8 --------------
#pragma unroll
  for (int i = 0; i < 2; ++i)
#pragma unroll
    for (int j = 0; j < 2; ++j)
#pragma unroll
      for (int r = 0; r < 8; ++r) {
        int row = m0 + wm + i * 16 + r + hi * 8;
        int col = n0 + wn + j * 16 + ln;
        float v = acc[i][j][r];
        if (bias) v += bias[col];
        if (add)  v += add[(size_t)row * N + col];
        if (act == 1)      v = fmaxf(v, 0.0f);
        else if (act == 2) v = dgelu(v);
        C[(size_t)row * N + col] = v;
      }
}

// ===========================================================================
// Flash attention: per block (b,h, 128 q rows); 8 waves x 16 q rows each.
// Streams 32 keys/step through shared K (row-major) and V (transposed) tiles,
// with next step's K/V global loads register-staged over this step's compute.
// ===========================================================================
__global__ void __launch_bounds__(256)
flash_attn_kernel(const float* __restrict__ Q, const float* __restrict__ Kb,
                  const float* __restrict__ Vb, float* __restrict__ O)
{
  __shared__ _Float16 sK[32 * 72];        // [key][dh]     (stride 72)
  __shared__ _Float16 sV[64 * LDP];       // [dh][key]     (transposed)
  __shared__ _Float16 sP[8 * 16 * LDP];   // per-wave P scratch 16x32

  const int tid = threadIdx.x, lane = tid & 31, wave = tid >> 5;
  const int hi = lane >> 4, ln = lane & 15;
  const int blk = blockIdx.x;             // b*128 + h*16 + qc
  const int qc = blk & 15, h = (blk >> 4) & 7, b = blk >> 7;
  const int q0 = qc * 128 + wave * 16;
  const size_t base = ((size_t)b * LL) * DD + h * DH;
  const float scale = 0.125f;             // 1/sqrt(64)

  // Q fragments (16 rows x 64 dh = two K=32 A-frags), loaded from global.
  v16h qf[2];
  {
    const float* qr = Q + base + (size_t)(q0 + ln) * DD;
#pragma unroll
    for (int ks = 0; ks < 2; ++ks) {
      float4 f0 = *(const float4*)(qr + ks * 32 + hi * 8);
      float4 f1 = *(const float4*)(qr + ks * 32 + hi * 8 + 4);
      float4 f2 = *(const float4*)(qr + ks * 32 + 16 + hi * 8);
      float4 f3 = *(const float4*)(qr + ks * 32 + 16 + hi * 8 + 4);
      v16h t;
      t[0]=(_Float16)f0.x; t[1]=(_Float16)f0.y; t[2]=(_Float16)f0.z; t[3]=(_Float16)f0.w;
      t[4]=(_Float16)f1.x; t[5]=(_Float16)f1.y; t[6]=(_Float16)f1.z; t[7]=(_Float16)f1.w;
      t[8]=(_Float16)f2.x; t[9]=(_Float16)f2.y; t[10]=(_Float16)f2.z; t[11]=(_Float16)f2.w;
      t[12]=(_Float16)f3.x; t[13]=(_Float16)f3.y; t[14]=(_Float16)f3.z; t[15]=(_Float16)f3.w;
      qf[ks] = t;
    }
  }

  float mrow[8], lrow[8];
  v8f oa[4] = {};
#pragma unroll
  for (int r = 0; r < 8; ++r) { mrow[r] = -1e30f; lrow[r] = 0.0f; }
  _Float16* myP = &sP[wave * 16 * LDP];

  // register staging for K/V tiles (32 keys x 64 dh each)
  float4 rk[2], rv[2];
  auto load_kv = [&](int kt) {
#pragma unroll
    for (int r = 0; r < 2; ++r) {
      int i = tid + 256 * r, key = i >> 4, c4 = i & 15;
      rk[r] = *(const float4*)(Kb + base + (size_t)(kt + key) * DD + c4 * 4);
      rv[r] = *(const float4*)(Vb + base + (size_t)(kt + key) * DD + c4 * 4);
    }
  };

  load_kv(0);
  for (int kt = 0; kt < LL; kt += 32) {
    __syncthreads();
#pragma unroll
    for (int r = 0; r < 2; ++r) {
      int i = tid + 256 * r, key = i >> 4, c4 = i & 15;
      _Float16* d = &sK[key * 72 + c4 * 4];
      d[0]=(_Float16)rk[r].x; d[1]=(_Float16)rk[r].y;
      d[2]=(_Float16)rk[r].z; d[3]=(_Float16)rk[r].w;
      sV[(c4 * 4 + 0) * LDP + key] = (_Float16)rv[r].x;
      sV[(c4 * 4 + 1) * LDP + key] = (_Float16)rv[r].y;
      sV[(c4 * 4 + 2) * LDP + key] = (_Float16)rv[r].z;
      sV[(c4 * 4 + 3) * LDP + key] = (_Float16)rv[r].w;
    }
    __syncthreads();
    if (kt + 32 < LL) load_kv(kt + 32);   // overlaps with compute below

    // S = scale * Q @ K^T   (two 16x16 C tiles: keys 0-15 / 16-31)
    v8f s0 = {}, s1 = {};
#pragma unroll
    for (int ks = 0; ks < 2; ++ks) {
      v8h a0 = *(const v8h*)&sK[ln * 72 + ks * 32 + hi * 8];
      v8h a1 = *(const v8h*)&sK[ln * 72 + ks * 32 + 16 + hi * 8];
      v8h b0 = *(const v8h*)&sK[(16 + ln) * 72 + ks * 32 + hi * 8];
      v8h b1 = *(const v8h*)&sK[(16 + ln) * 72 + ks * 32 + 16 + hi * 8];
      s0 = __builtin_amdgcn_wmma_f32_16x16x32_f16(false, qf[ks], false, cat8(a0, a1), (short)0, s0, false, false);
      s1 = __builtin_amdgcn_wmma_f32_16x16x32_f16(false, qf[ks], false, cat8(b0, b1), (short)0, s1, false, false);
    }

    // online softmax update (row reductions across the 16-lane half)
#pragma unroll
    for (int r = 0; r < 8; ++r) {
      float a = s0[r] * scale, c = s1[r] * scale;
      float rm = fmaxf(a, c);
      rm = fmaxf(rm, __shfl_xor(rm, 1, 32));
      rm = fmaxf(rm, __shfl_xor(rm, 2, 32));
      rm = fmaxf(rm, __shfl_xor(rm, 4, 32));
      rm = fmaxf(rm, __shfl_xor(rm, 8, 32));
      float mn  = fmaxf(mrow[r], rm);
      float cor = __expf(mrow[r] - mn);
      float p0  = __expf(a - mn), p1 = __expf(c - mn);
      float ps  = p0 + p1;
      ps += __shfl_xor(ps, 1, 32);
      ps += __shfl_xor(ps, 2, 32);
      ps += __shfl_xor(ps, 4, 32);
      ps += __shfl_xor(ps, 8, 32);
      lrow[r] = lrow[r] * cor + ps;
      mrow[r] = mn;
#pragma unroll
      for (int nh = 0; nh < 4; ++nh) oa[nh][r] *= cor;
      myP[(r + hi * 8) * LDP + ln]      = (_Float16)p0;   // C layout -> LDS
      myP[(r + hi * 8) * LDP + 16 + ln] = (_Float16)p1;
    }
    asm volatile("s_wait_dscnt 0x0" ::: "memory");        // LDS relayout fence

    // P fragment (A layout) + O += P @ V
    v8h p0 = *(const v8h*)&myP[ln * LDP + hi * 8];
    v8h p1 = *(const v8h*)&myP[ln * LDP + 16 + hi * 8];
    v16h pf = cat8(p0, p1);
#pragma unroll
    for (int nh = 0; nh < 4; ++nh) {
      v8h v0 = *(const v8h*)&sV[(nh * 16 + ln) * LDP + hi * 8];
      v8h v1 = *(const v8h*)&sV[(nh * 16 + ln) * LDP + 16 + hi * 8];
      oa[nh] = __builtin_amdgcn_wmma_f32_16x16x32_f16(false, pf, false, cat8(v0, v1), (short)0, oa[nh], false, false);
    }
  }

#pragma unroll
  for (int nh = 0; nh < 4; ++nh)
#pragma unroll
    for (int r = 0; r < 8; ++r) {
      int row = q0 + r + hi * 8;
      O[base + (size_t)row * DD + nh * 16 + ln] = oa[nh][r] / lrow[r];
    }
}

// ===========================================================================
// Elementwise / scan kernels
// ===========================================================================
__global__ void diff_pad_kernel(const float* __restrict__ x, float* __restrict__ out) {
  size_t idx = (size_t)blockIdx.x * 256 + threadIdx.x;      // BLD exact
  int d = idx % DD; size_t r = idx / DD; int l = r % LL; size_t b = r / LL;
  size_t o = ((size_t)b * LL + l) * DD + d;
  out[o] = (l == 0) ? x[(b * LL + 1) * DD + d] - x[(b * LL) * DD + d]
                    : x[o] - x[o - DD];
}

__global__ void ema_fused_kernel(const float* __restrict__ vel,
                                 const float* __restrict__ logits,
                                 float* __restrict__ fused) {
  int t = blockIdx.x * 256 + threadIdx.x;   // B*D = 2048 threads
  int b = t >> 9, d = t & 511;
  float a0 = dsigmoid(logits[0]), a1 = dsigmoid(logits[1]), a2 = dsigmoid(logits[2]);
  float e0 = __expf(a0), e1 = __expf(a1), e2 = __expf(a2);
  float inv = 1.0f / (e0 + e1 + e2);
  float w0 = e0 * inv, w1 = e1 * inv, w2 = e2 * inv;
  size_t base = (size_t)b * LL * DD + d;
  float v = vel[base];
  float m0 = v, m1 = v, m2 = v;
  fused[base] = w0 * m0 + w1 * m1 + w2 * m2;
  for (int l = 1; l < LL; ++l) {
    v = vel[base + (size_t)l * DD];
    m0 = a0 * m0 + (1.0f - a0) * v;
    m1 = a1 * m1 + (1.0f - a1) * v;
    m2 = a2 * m2 + (1.0f - a2) * v;
    fused[base + (size_t)l * DD] = w0 * m0 + w1 * m1 + w2 * m2;
  }
}

__global__ void cumsum_trend_kernel(const float* __restrict__ traw,
                                    const float* __restrict__ xin,
                                    float* __restrict__ trend) {
  int t = blockIdx.x * 256 + threadIdx.x;   // B*D threads
  int b = t >> 9, d = t & 511;
  size_t base = (size_t)b * LL * DD + d;
  float anchor = xin[base] - traw[base];    // x[:,0] - tc[:,0]  (tc0 = traw0)
  float run = 0.0f;
  for (int l = 0; l < LL; ++l) {
    run += traw[base + (size_t)l * DD];
    trend[base + (size_t)l * DD] = run + anchor;
  }
}

__global__ void dwconv5_kernel(const float* __restrict__ trend,
                               const float* __restrict__ k5,
                               const float* xin, float* season, float* trend_s) {
  size_t idx = (size_t)blockIdx.x * 256 + threadIdx.x;      // BLD exact
  int d = idx % DD; size_t r = idx / DD; int l = r % LL; size_t b = r / LL;
  size_t base = ((size_t)b * LL) * DD + d;
  float acc = 0.0f;
#pragma unroll
  for (int j = 0; j < 5; ++j) {
    int ll = l + j - 2;
    if (ll >= 0 && ll < LL) acc += trend[base + (size_t)ll * DD] * k5[d * 5 + j];
  }
  trend_s[idx] = acc;
  season[idx]  = xin[idx] - acc;
}

__global__ void __launch_bounds__(256)
ln_gelu_kernel(const float* X, const float* __restrict__ g,
               const float* __restrict__ be, float* Y) {
  __shared__ float red[256];
  int row = blockIdx.x, tid = threadIdx.x;
  size_t base = (size_t)row * DD;
  float v0 = X[base + tid], v1 = X[base + 256 + tid];
  red[tid] = v0 + v1; __syncthreads();
  for (int s = 128; s > 0; s >>= 1) { if (tid < s) red[tid] += red[tid + s]; __syncthreads(); }
  float mean = red[0] / (float)DD; __syncthreads();
  float d0 = v0 - mean, d1 = v1 - mean;
  red[tid] = d0 * d0 + d1 * d1; __syncthreads();
  for (int s = 128; s > 0; s >>= 1) { if (tid < s) red[tid] += red[tid + s]; __syncthreads(); }
  float rstd = rsqrtf(red[0] / (float)DD + 1e-5f);
  Y[base + tid]       = dgelu(d0 * rstd * g[tid] + be[tid]);
  Y[base + 256 + tid] = dgelu(d1 * rstd * g[tid + 256] + be[tid + 256]);
}

__global__ void __launch_bounds__(256)
rowdot_sigmoid_kernel(const float* __restrict__ Hf, const float* __restrict__ w,
                      const float* __restrict__ b0, float* __restrict__ out) {
  int lane = threadIdx.x & 31, wave = threadIdx.x >> 5;
  int row = blockIdx.x * 8 + wave;
  size_t base = (size_t)row * 256;
  float s = 0.0f;
#pragma unroll
  for (int i = 0; i < 8; ++i) s += Hf[base + lane + 32 * i] * w[lane + 32 * i];
  s += __shfl_xor(s, 16, 32); s += __shfl_xor(s, 8, 32);
  s += __shfl_xor(s, 4, 32);  s += __shfl_xor(s, 2, 32); s += __shfl_xor(s, 1, 32);
  if (lane == 0) out[row] = dsigmoid(s + b0[0]);
}

__global__ void mom_combine_kernel(const float* tc, const float* __restrict__ proj,
                                   const float* __restrict__ aw, const float* __restrict__ fw,
                                   const float* __restrict__ gws, const float* gate,
                                   float* out, int useGate) {
  size_t idx = (size_t)blockIdx.x * 256 + threadIdx.x;      // BLD exact
  int d = idx % DD; size_t row = idx / DD;
  float m = proj[idx] * (dsigmoid(gws[0]) * aw[row]) * dsigmoid(fw[d]);
  if (useGate) m *= dsigmoid(gate[idx]);
  out[idx] = tc[idx] + m;
}

__global__ void sub_kernel(const float* a, const float* b, float* o) {
  size_t i = (size_t)blockIdx.x * 256 + threadIdx.x; o[i] = a[i] - b[i];
}
__global__ void add3_kernel(const float* a, const float* b, const float* c, float* o) {
  size_t i = (size_t)blockIdx.x * 256 + threadIdx.x; o[i] = a[i] + b[i] + c[i];
}

__global__ void gatherG_kernel(const float* __restrict__ ts, float* __restrict__ G) {
  size_t idx = (size_t)blockIdx.x * 256 + threadIdx.x;      // B*L*3*D exact
  int ci = idx % DD; size_t r = idx / DD; int t = r % 3; r /= 3;
  int l = r % LL; int b = (int)(r / LL);
  int ls = l - 1 + t; if (ls < 0) ls += LL; if (ls >= LL) ls -= LL;
  G[idx] = ts[((size_t)b * LL + ls) * DD + ci];
}

__global__ void wtrans_kernel(const float* __restrict__ pk, float* __restrict__ Wc) {
  size_t idx = (size_t)blockIdx.x * 256 + threadIdx.x;      // 1536*512 exact
  int co = idx % DD; size_t kidx = idx / DD;
  int ci = kidx % DD; int t = (int)(kidx / DD);
  Wc[idx] = pk[((size_t)co * DD + ci) * 3 + t];
}

// ===========================================================================
// Host orchestration
// ===========================================================================
extern "C" void kernel_launch(void* const* d_in, const int* in_sizes, int n_in,
                              void* d_out, int out_size, void* d_ws, size_t ws_size,
                              hipStream_t stream) {
  (void)in_sizes; (void)n_in; (void)out_size; (void)ws_size;
  auto F = [&](int i) -> const float* { return (const float*)d_in[i]; };

  float* ws = (float*)d_ws;
  const size_t S = BLD;
  float* qb = ws + 0 * S; float* kb = ws + 1 * S; float* vb = ws + 2 * S; float* cx = ws + 3 * S;
  float* xc = ws + 4 * S; float* t1 = ws + 5 * S; float* t2 = ws + 6 * S; float* t3 = ws + 7 * S;
  float* s0 = ws + 8 * S; float* s1 = ws + 9 * S; float* s2 = ws + 10 * S; float* s3 = ws + 11 * S;
  float* awh = ws + 12 * S; float* awv = awh + (size_t)BL * 256;
  float* hff = ws;          // reuses slabs 0-3 (free during FFN)
  float* G   = ws + 1 * S;  // reuses slabs 1-3 (free at final conv)
  float* Wc  = ws + 3 * S;  // reuses slab 3

  const int ewb = (int)(BLD / 256);

  auto gemm = [&](const float* A, const float* W, const float* bias, const float* add,
                  float* C, int M, int N, int K, int act) {
    dim3 grid(N / 64, M / 64);
    gemm_wmma_kernel<<<grid, 128, 0, stream>>>(A, W, bias, add, C, M, N, K, act);
  };

  auto mha = [&](const float* qin, const float* kvin, int pb, const float* resid, float* xout) {
    gemm(qin,  F(pb + 0), nullptr, nullptr, qb, BL, DD, DD, 0);
    gemm(kvin, F(pb + 1), nullptr, nullptr, kb, BL, DD, DD, 0);
    gemm(kvin, F(pb + 2), nullptr, nullptr, vb, BL, DD, DD, 0);
    flash_attn_kernel<<<BB * HH * (LL / 128), 256, 0, stream>>>(qb, kb, vb, cx);
    gemm(cx, F(pb + 3), nullptr, resid, xout, BL, DD, DD, 0);
  };

  auto decomp = [&](int pb, float* xio, float* tout, float* season) {
    diff_pad_kernel<<<ewb, 256, 0, stream>>>(xio, s0);
    ema_fused_kernel<<<8, 256, 0, stream>>>(s0, F(pb + 0), s1);
    gemm(s1, F(pb + 1), F(pb + 2), nullptr, s2, BL, DD, DD, 0);
    ln_gelu_kernel<<<BL, 256, 0, stream>>>(s2, F(pb + 3), F(pb + 4), s2);
    gemm(s2, F(pb + 5), F(pb + 6), nullptr, s0, BL, DD, DD, 0);
    cumsum_trend_kernel<<<8, 256, 0, stream>>>(s0, xio, s3);
    dwconv5_kernel<<<ewb, 256, 0, stream>>>(s3, F(pb + 7), xio, season, tout);
  };

  auto momentum = [&](int mb, float* t, const float* tp) {
    if (tp) sub_kernel<<<ewb, 256, 0, stream>>>(t, tp, s0);
    else    diff_pad_kernel<<<ewb, 256, 0, stream>>>(t, s0);
    gemm(s0, F(mb + 6), F(mb + 7), nullptr, s1, BL, DD, DD, 0);        // pw1+pb1
    ln_gelu_kernel<<<BL, 256, 0, stream>>>(s1, F(mb + 8), F(mb + 9), s1);
    gemm(s1, F(mb + 10), F(mb + 11), nullptr, s2, BL, DD, DD, 0);      // pw2+pb2 -> proj
    gemm(t, F(mb + 2), F(mb + 3), nullptr, awh, BL, 256, DD, 2);       // gelu(tc@aw1+ab1)
    rowdot_sigmoid_kernel<<<BL / 8, 256, 0, stream>>>(awh, F(mb + 4), F(mb + 5), awv);
    if (tp) {  // gate = sigmoid([tc, proj] @ gw + gb)  (split K)
      gemm(t,  F(mb + 12), F(mb + 13), nullptr, s3, BL, DD, DD, 0);
      gemm(s2, F(mb + 12) + (size_t)DD * DD, nullptr, s3, s3, BL, DD, DD, 0);
    }
    mom_combine_kernel<<<ewb, 256, 0, stream>>>(t, s2, awv, F(mb + 1), F(mb + 0), s3, t, tp ? 1 : 0);
  };

  const float* x     = F(0);
  const float* cross = F(1);
  float* outx = (float*)d_out;
  float* outt = outx + BLD;

  // x = x + self_mha(x,x)
  mha(x, x, 2, x, xc);
  // decomp1 + momentum1
  decomp(10, xc, t1, xc);
  momentum(34, t1, nullptr);
  // x = x + cross_mha(x, cross)
  mha(xc, cross, 6, xc, xc);
  // decomp2 + momentum2
  decomp(18, xc, t2, xc);
  momentum(48, t2, t1);
  // FFN: x = x + relu(x@w1)@w2
  gemm(xc, F(76), nullptr, nullptr, hff, BL, DFF, DD, 1);
  gemm(hff, F(77), nullptr, xc, xc, BL, DD, DFF, 0);
  // decomp3 (season -> output slot 0) + momentum3
  decomp(26, xc, t3, outx);
  momentum(62, t3, t2);
  // residual_trend = circ_conv3(t1+t2+t3)  == gather + K=1536 WMMA GEMM
  add3_kernel<<<ewb, 256, 0, stream>>>(t1, t2, t3, s0);
  gatherG_kernel<<<(int)((size_t)BL * 3 * DD / 256), 256, 0, stream>>>(s0, G);
  wtrans_kernel<<<(int)((size_t)3 * DD * DD / 256), 256, 0, stream>>>(F(78), Wc);
  gemm(G, Wc, nullptr, nullptr, outt, BL, DD, 3 * DD, 0);
}
